// TemporalLightGNN_2499670966899
// MI455X (gfx1250) — matrix operand order repaired and checked by reading, not compiled, not run on hardware
//
#include <hip/hip_runtime.h>
#include <math.h>

// ---------------- problem constants ----------------
#define BB      256          // batch
#define LL      256          // sequence length
#define DM      256          // d_model
#define ED      512          // embed_dim
#define NNODES  (BB + BB*LL) // 65792 graph nodes (users + events)
#define LN_EPS  1e-5f

typedef __attribute__((ext_vector_type(16))) __bf16 v16bf;
typedef __attribute__((ext_vector_type(8)))  __bf16 v8bf;
typedef __attribute__((ext_vector_type(8)))  float  v8f;
typedef __attribute__((address_space(3))) __bf16 lds_bf16;

// analytic symmetric-norm factors of the fixed graph
#define DIS_U   0.06237828615518053f   // rsqrt(257): user node, deg = L + self = 257
#define DIS_E3  0.5773502691896258f    // rsqrt(3): chain-endpoint event
#define DIS_E4  0.5f                   // rsqrt(4): interior event
__device__ __forceinline__ float dis_ev(int l) {
    return (l == 0 || l == LL - 1) ? DIS_E3 : DIS_E4;
}

// ---------------- utility kernels ----------------
__global__ void zero_node(float* x, __bf16* xb, int n) {
    int i = blockIdx.x * blockDim.x + threadIdx.x;
    if (i < n) { x[i] = 0.f; xb[i] = (__bf16)0.f; }
}

__global__ void f32_to_bf16(const float* __restrict__ src, __bf16* __restrict__ dst, int n) {
    for (int i = blockIdx.x * blockDim.x + threadIdx.x; i < n; i += gridDim.x * blockDim.x)
        dst[i] = (__bf16)src[i];
}

// ---------------- stage 1: fused embed + proj + GELU + LN + PE (dual f32/bf16 output) ----
// one block per (b,l) row, 256 threads = 256 features
__global__ void embed_kernel(const int* __restrict__ type_ids, const int* __restrict__ sku_ids,
                             const int* __restrict__ cat_ids,  const int* __restrict__ price_ids,
                             const int* __restrict__ url_ids,  const float* __restrict__ qv,
                             const float* __restrict__ e_type, const float* __restrict__ e_sku,
                             const float* __restrict__ e_cat,  const float* __restrict__ e_url,
                             const float* __restrict__ e_price,
                             const float* __restrict__ qW, const float* __restrict__ qb,
                             const float* __restrict__ ln_w, const float* __restrict__ ln_b,
                             float* __restrict__ X, __bf16* __restrict__ XB) {
    const int idx = blockIdx.x;          // b*L + l
    const int l   = idx & (LL - 1);
    const int d   = threadIdx.x;

    const int ti = type_ids[idx]  & 7;
    const int si = sku_ids[idx]   & 65535;
    const int ci = cat_ids[idx]   & 4095;
    const int pi = price_ids[idx] & 127;
    const int ui = url_ids[idx]   & 65535;

    float s = e_type[ti*DM + d] + e_sku[si*DM + d] + e_cat[ci*DM + d] +
              e_price[pi*DM + d] + e_url[ui*DM + d];

    __shared__ float q[16];
    if (d < 16) q[d] = qv[(size_t)idx * 16 + d];
    __syncthreads();
#pragma unroll
    for (int j = 0; j < 16; ++j) s += q[j] * qW[d*16 + j];
    s += qb[d];

    // exact GELU
    s = 0.5f * s * (1.0f + erff(s * 0.7071067811865476f));

    // LayerNorm over 256 features
    __shared__ float red[DM];
    red[d] = s; __syncthreads();
    for (int st = DM/2; st > 0; st >>= 1) {
        if (d < st) red[d] += red[d + st];
        __syncthreads();
    }
    const float mu = red[0] * (1.0f/DM);
    __syncthreads();
    float dv = s - mu;
    red[d] = dv * dv; __syncthreads();
    for (int st = DM/2; st > 0; st >>= 1) {
        if (d < st) red[d] += red[d + st];
        __syncthreads();
    }
    const float var = red[0] * (1.0f/DM);
    float xn = dv * rsqrtf(var + LN_EPS) * ln_w[d] + ln_b[d];

    // sinusoidal positional encoding
    const float freq = __expf((float)(d & ~1) * (-9.210340371976184f / (float)DM));
    const float ang  = (float)l * freq;
    xn += (d & 1) ? __cosf(ang) : __sinf(ang);

    const size_t o = (size_t)(BB + idx) * DM + d;   // event node row
    X[o]  = xn;
    XB[o] = (__bf16)xn;
}

// ---------------- bf16 WMMA GEMM: C[M,N] = A[M,K] * W[N,K]^T + bias, opt ReLU ----------
// block = 8 waves sharing one tm (16 A-rows): A tile async-staged to LDS once,
// each wave owns one 16x16 C tile (tn = 8 consecutive tiles per block).
// Requires: M%16==0, N%128==0, K%32==0, K*2*16 bytes <= LDS budget (K=256 -> 8KB).
__global__ void gemm_bf16_wmma(const __bf16* __restrict__ A, const __bf16* __restrict__ W,
                               const float* __restrict__ bias, float* __restrict__ C,
                               int M, int K, int N, int act) {
    __shared__ __bf16 sA[16 * 256];                 // 16 rows x K=256, 8KB
    const int lane = threadIdx.x & 31;
    const int wid  = threadIdx.x >> 5;
    const int bpr  = N >> 7;                        // blocks per tm row (tnc/8)
    const int tm   = blockIdx.x / bpr;
    const int tn   = (blockIdx.x % bpr) * 8 + wid;

    // ---- async-stage the contiguous 16xK A tile into LDS (all 256 threads) ----
    const uint32_t lds_base = (uint32_t)(uintptr_t)(lds_bf16*)&sA[0];
    const uint64_t gbase    = (uint64_t)(const void*)(A + (size_t)(tm << 4) * K);
    const int nChunk = 2 * K;                        // 16*K*2 bytes / 16B chunks
    for (int c = threadIdx.x; c < nChunk; c += 256) {
        uint32_t loff = lds_base + (uint32_t)c * 16;
        uint64_t g    = gbase + (uint64_t)c * 16;
        asm volatile("global_load_async_to_lds_b128 %0, %1, off"
                     :: "v"(loff), "v"(g) : "memory");
    }
    asm volatile("s_wait_asynccnt 0" ::: "memory");
    __syncthreads();

    const int ncol  = (tn << 4) + (lane & 15);
    const int koffA = (lane < 16) ? 0 : 16;
    const int koffB = (lane < 16) ? 0 : 16;
    const __bf16* arow = &sA[(lane & 15) * K];      // LDS, ds_load_b128
    const __bf16* wrow = W + (size_t)ncol * K;

    v8f c = {};
    for (int k0 = 0; k0 < K; k0 += 32) {
        v8bf alo = *(const v8bf*)(arow + k0 + koffA);
        v8bf ahi = *(const v8bf*)(arow + k0 + koffA + 8);
        v16bf a;
#pragma unroll
        for (int i = 0; i < 8; ++i) { a[i] = alo[i]; a[i + 8] = ahi[i]; }
        v16bf b = *(const v16bf*)(wrow + k0 + koffB);
        c = __builtin_amdgcn_wmma_f32_16x16x32_bf16(false, a, false, b, (short)0, c, false, false);
    }

    const float bv   = bias[ncol];
    const int  mbase = (tm << 4) + ((lane >> 4) << 3);
#pragma unroll
    for (int r = 0; r < 8; ++r) {
        float v = c[r] + bv;
        if (act) v = fmaxf(v, 0.f);
        C[(size_t)(mbase + r) * N + ncol] = v;
    }
}

// ---------------- message passing (analytic graph stencil), dual f32/bf16 output -------
// event rows: X_ev(b,l) = relu( d_l * (d_u*H_u(b) + d_{l-1}H(b,l-1) + d_{l+1}H(b,l+1) + d_l H(b,l)) )
__global__ void msg_events(const float* __restrict__ H, float* __restrict__ X,
                           __bf16* __restrict__ XB) {
    const int idx = blockIdx.x;           // b*L + l
    const int l   = idx & (LL - 1);
    const int b   = idx >> 8;
    const int d   = threadIdx.x;
    const float dl = dis_ev(l);
    float acc = DIS_U * H[(size_t)b * DM + d];            // user node row = b
    acc += dl * H[(size_t)(BB + idx) * DM + d];           // self
    if (l > 0)      acc += dis_ev(l - 1) * H[(size_t)(BB + idx - 1) * DM + d];
    if (l < LL - 1) acc += dis_ev(l + 1) * H[(size_t)(BB + idx + 1) * DM + d];
    const float v = fmaxf(dl * acc, 0.f);
    const size_t o = (size_t)(BB + idx) * DM + d;
    X[o]  = v;
    XB[o] = (__bf16)v;
}

// user rows: X_u(b) = relu( d_u * sum_l d_l H_ev(b,l) + d_u^2 H_u(b) )
__global__ void msg_users(const float* __restrict__ H, float* __restrict__ X,
                          __bf16* __restrict__ XB) {
    const int b = blockIdx.x;
    const int d = threadIdx.x;
    float acc = 0.f;
    const float* Hev = H + (size_t)(BB + b * LL) * DM + d;
    for (int l = 0; l < LL; ++l)
        acc += dis_ev(l) * Hev[(size_t)l * DM];
    float v = DIS_U * acc + DIS_U * DIS_U * H[(size_t)b * DM + d];
    v = fmaxf(v, 0.f);
    const size_t o = (size_t)b * DM + d;
    X[o]  = v;
    XB[o] = (__bf16)v;
}

// ---------------- logits: one wave per event row, 8 dot products ----------------
__global__ void logits_kernel(const float* __restrict__ X, const float* __restrict__ hW,
                              const float* __restrict__ hb, float* __restrict__ out) {
    const int wave = (blockIdx.x * blockDim.x + threadIdx.x) >> 5;
    const int lane = threadIdx.x & 31;
    if (wave >= BB * LL) return;
    const float* ev = X + (size_t)(BB + wave) * DM;
    float e[8];
#pragma unroll
    for (int j = 0; j < 8; ++j) e[j] = ev[lane + 32 * j];
#pragma unroll
    for (int t = 0; t < 8; ++t) {
        float p = 0.f;
#pragma unroll
        for (int j = 0; j < 8; ++j) p += e[j] * hW[t * DM + lane + 32 * j];
        for (int off = 16; off > 0; off >>= 1) p += __shfl_xor(p, off, 32);
        if (lane == 0) out[(size_t)wave * 8 + t] = p + hb[t];
    }
}

// ---------------- row L2 normalize (512-wide rows) ----------------
__global__ void normalize_kernel(const float* __restrict__ U, float* __restrict__ out) {
    const int row = blockIdx.x, d = threadIdx.x;
    const float v0 = U[(size_t)row * ED + d];
    const float v1 = U[(size_t)row * ED + d + DM];
    __shared__ float red[DM];
    red[d] = v0 * v0 + v1 * v1; __syncthreads();
    for (int st = DM/2; st > 0; st >>= 1) {
        if (d < st) red[d] += red[d + st];
        __syncthreads();
    }
    const float s = 1.f / fmaxf(sqrtf(red[0]), 1e-12f);
    out[(size_t)row * ED + d]      = v0 * s;
    out[(size_t)row * ED + d + DM] = v1 * s;
}

// ---------------- launcher ----------------
extern "C" void kernel_launch(void* const* d_in, const int* in_sizes, int n_in,
                              void* d_out, int out_size, void* d_ws, size_t ws_size,
                              hipStream_t stream) {
    const int*   type_ids  = (const int*)  d_in[0];
    const int*   sku_ids   = (const int*)  d_in[1];
    const int*   cat_ids   = (const int*)  d_in[2];
    const int*   price_ids = (const int*)  d_in[3];
    const int*   url_ids   = (const int*)  d_in[4];
    const float* query_vec = (const float*)d_in[5];
    const float* emb_type  = (const float*)d_in[6];
    const float* emb_sku   = (const float*)d_in[7];
    const float* emb_cat   = (const float*)d_in[8];
    const float* emb_url   = (const float*)d_in[9];
    const float* emb_price = (const float*)d_in[10];
    const float* qW   = (const float*)d_in[11];
    const float* qb   = (const float*)d_in[12];
    const float* ln_w = (const float*)d_in[13];
    const float* ln_b = (const float*)d_in[14];
    const float* Ws[3] = { (const float*)d_in[15], (const float*)d_in[17], (const float*)d_in[19] };
    const float* bs[3] = { (const float*)d_in[16], (const float*)d_in[18], (const float*)d_in[20] };
    const float* upW1 = (const float*)d_in[21];
    const float* upb1 = (const float*)d_in[22];
    const float* upW2 = (const float*)d_in[23];
    const float* upb2 = (const float*)d_in[24];
    const float* hW   = (const float*)d_in[25];
    const float* hb   = (const float*)d_in[26];
    float* out = (float*)d_out;   // [u: 256*512][logits: 256*256*8]

    // workspace layout
    const size_t SZ_X  = (size_t)NNODES * DM * sizeof(float);   // 67,371,008
    const size_t SZ_XB = (size_t)NNODES * DM * sizeof(__bf16);  // 33,685,504
    char* ws = (char*)d_ws;
    float*  X   = (float*)(ws);
    float*  H   = (float*)(ws + SZ_X);
    __bf16* XB  = (__bf16*)(ws + 2 * SZ_X);
    __bf16* WB  = (__bf16*)(ws + 2 * SZ_X + SZ_XB);             // up to 512*256 bf16
    __bf16* U1B = (__bf16*)(ws + 2 * SZ_X + SZ_XB + (size_t)ED * DM * sizeof(__bf16));
    float*  U1  = H;                                            // 256x256, H free at head stage
    float*  U2  = H + (1 << 18);                                // 256x512, disjoint from U1

    // 1) node features: users = 0, events = fused embed pipeline (dual f32 + bf16)
    zero_node<<<(BB * DM + 255) / 256, 256, 0, stream>>>(X, XB, BB * DM);
    embed_kernel<<<BB * LL, DM, 0, stream>>>(type_ids, sku_ids, cat_ids, price_ids, url_ids,
                                             query_vec, emb_type, emb_sku, emb_cat, emb_url,
                                             emb_price, qW, qb, ln_w, ln_b, X, XB);

    // 2) three GCN layers: H = X*W^T + b (bf16 WMMA, LDS-staged A), analytic message pass
    const int gcnBlocks = (NNODES >> 4) * (DM >> 7);            // (M/16) * (N/128)
    for (int layer = 0; layer < 3; ++layer) {
        f32_to_bf16<<<256, 256, 0, stream>>>(Ws[layer], WB, DM * DM);
        gemm_bf16_wmma<<<gcnBlocks, 256, 0, stream>>>(XB, WB, bs[layer], H, NNODES, DM, DM, 0);
        msg_users<<<BB, DM, 0, stream>>>(H, X, XB);
        msg_events<<<BB * LL, DM, 0, stream>>>(H, X, XB);
    }

    // 3) user head: U1 = relu(user @ upW1^T + b), U2 = U1 @ upW2^T + b, L2-normalize
    f32_to_bf16<<<256, 256, 0, stream>>>(upW1, WB, DM * DM);
    gemm_bf16_wmma<<<(BB >> 4) * (DM >> 7), 256, 0, stream>>>(XB, WB, upb1, U1, BB, DM, DM, 1);
    f32_to_bf16<<<256, 256, 0, stream>>>(U1, U1B, BB * DM);
    f32_to_bf16<<<512, 256, 0, stream>>>(upW2, WB, ED * DM);
    gemm_bf16_wmma<<<(BB >> 4) * (ED >> 7), 256, 0, stream>>>(U1B, WB, upb2, U2, BB, DM, ED, 0);
    normalize_kernel<<<BB, DM, 0, stream>>>(U2, out);

    // 4) logits from final event features
    logits_kernel<<<(BB * LL) / 8, 256, 0, stream>>>(X, hW, hb, out + (size_t)BB * ED);
}
